// class_network_18743237280674
// MI455X (gfx1250) — matrix operand order repaired and checked by reading, not compiled, not run on hardware
//
#include <hip/hip_runtime.h>
#include <hip/hip_bf16.h>

// Problem constants (match reference)
#define BB   64
#define INC  64
#define LL   4096
#define CC   128
#define DD   192
#define NCLS 1000
#define NE   8
#define EPSV 1e-5f

typedef __attribute__((ext_vector_type(16))) __bf16 bf16x16;
typedef __attribute__((ext_vector_type(8)))  float  floatx8;

union FragAB { bf16x16 v; uint4 q[2]; };

__device__ __forceinline__ float bf2f(unsigned short u) {
  unsigned int x = ((unsigned int)u) << 16;
  float f; __builtin_memcpy(&f, &x, 4); return f;
}
__device__ __forceinline__ unsigned short f2bf(float f) {
  unsigned int x; __builtin_memcpy(&x, &f, 4);
  unsigned int lsb = (x >> 16) & 1u;
  x += 0x7fffu + lsb;                 // round-to-nearest-even
  return (unsigned short)(x >> 16);
}

// ---------------- gfx1250 async global->LDS path (guarded) ----------------
#if defined(__has_builtin)
#if __has_builtin(__builtin_amdgcn_global_load_async_to_lds_b128)
#define HAVE_ASYNC_LDS 1
#endif
#endif
#ifndef HAVE_ASYNC_LDS
#define HAVE_ASYNC_LDS 0
#endif

#if HAVE_ASYNC_LDS
typedef __attribute__((ext_vector_type(4))) int v4i_t;
typedef __attribute__((address_space(1))) v4i_t gbl_v4i;
typedef __attribute__((address_space(3))) v4i_t lds_v4i;
// NOTE: on amdgcn a generic pointer to LDS carries the DS byte offset in its
// low 32 bits, so an integer round-trip yields a valid AS(3) pointer; global
// generic pointers share their bit pattern with AS(1).
__device__ __forceinline__ void async_cp16(const void* g, const void* l) {
  __builtin_amdgcn_global_load_async_to_lds_b128(
      (gbl_v4i*)(unsigned long long)(uintptr_t)g,
      (lds_v4i*)(unsigned int)(uintptr_t)l, 0, 0);
}
__device__ __forceinline__ void async_wait0() {
#if __has_builtin(__builtin_amdgcn_s_wait_asynccnt)
  __builtin_amdgcn_s_wait_asynccnt(0);
#else
  asm volatile("s_wait_asynccnt 0x0" ::: "memory");
#endif
}
#endif

// ---------------- x: [B,IN,L] f32 -> [IN,B,L] bf16 ----------------
__global__ void convert_x_kernel(const float* __restrict__ x,
                                 unsigned short* __restrict__ xb) {
  int row = blockIdx.x;                 // row = b*INC + ci
  int b = row / INC, ci = row % INC;
  size_t src = (size_t)row * LL;
  size_t dst = ((size_t)ci * BB + b) * LL;
  for (int i = threadIdx.x; i < LL; i += blockDim.x)
    xb[dst + i] = f2bf(x[src + i]);
}

// ------------- weights: [Cout,Cin,ks] f32 -> [Cout][t*Cin+ci] bf16 -------------
__global__ void repack_w_kernel(const float* __restrict__ w,
                                unsigned short* __restrict__ wp,
                                int Cout, int Cin, int ks) {
  int idx = blockIdx.x * blockDim.x + threadIdx.x;   // output index
  int tot = Cout * Cin * ks;
  if (idx >= tot) return;
  int co  = idx / (Cin * ks);
  int rem = idx % (Cin * ks);
  int t   = rem / Cin;
  int ci  = rem % Cin;
  wp[idx] = f2bf(w[((size_t)co * Cin + ci) * ks + t]);
}

// ---------------- conv1d as implicit GEMM with bf16 WMMA ----------------
// in:  [Cin][B][L] bf16 raw (optionally normalize+relu on load)
// wp:  [Cout][ks*Cin] bf16  (K = t*Cin+ci)
// out: [Cout][B][L] bf16 raw conv output (pre-BN)
// LDS tile staged ONCE per ci-chunk covering all taps: rows l0-pad .. l0+63+pad.
__global__ __launch_bounds__(128) void conv_wmma_kernel(
    const unsigned short* __restrict__ in, const unsigned short* __restrict__ wp,
    unsigned short* __restrict__ out,
    const float* __restrict__ sc, const float* __restrict__ sh, int use_act,
    int Cin, int Cout, int ks, int pad) {
  __shared__ __align__(16) unsigned short Bs[68][40];  // [n_ext][k=ci], 80B row stride

  int tile = blockIdx.x;                    // over B * L/64
  int b    = tile / (LL / 64);
  int l0   = (tile % (LL / 64)) * 64;
  int wave = threadIdx.x >> 5;
  int lane = threadIdx.x & 31;
  int co_base = blockIdx.y * 64 + wave * 16;
  int m   = lane & 15;
  int hlf = lane >> 4;
  int Kt  = Cin * ks;
  int nrows = 64 + ks - 1;                  // 2*pad == ks-1 for these convs

  floatx8 acc[4];
#pragma unroll
  for (int j = 0; j < 4; ++j) acc[j] = (floatx8)0.0f;

  int r = threadIdx.x >> 2;   // staging: ci row 0..31
  int q = threadIdx.x & 3;    // staging: column quarter

  for (int ci0 = 0; ci0 < Cin; ci0 += 32) {
    __syncthreads();   // protect LDS from previous-iteration readers
    // ---- stage B tile (transposed) into LDS, fusing BN+ReLU of prev layer ----
    {
      int ci = ci0 + r;
      size_t rowbase = ((size_t)ci * BB + b) * LL;
      float scv = use_act ? sc[ci] : 1.0f;
      float shv = use_act ? sh[ci] : 0.0f;
#pragma unroll
      for (int i = 0; i < 17; ++i) {
        int ne = q * 17 + i;
        if (ne < nrows) {
          int lg = l0 - pad + ne;
          unsigned short u = 0;
          if (lg >= 0 && lg < LL) u = in[rowbase + lg];
          if (use_act) u = f2bf(fmaxf(bf2f(u) * scv + shv, 0.0f));
          Bs[ne][r] = u;
        }
      }
      // warm L2/WGP$ for the next ci-chunk (global_prefetch_b8)
      if (ci0 + 32 < Cin)
        __builtin_prefetch(in + rowbase + (size_t)32 * BB * LL + l0 + q * 16, 0, 1);
    }
    __syncthreads();

    for (int t = 0; t < ks; ++t) {
      // ---- A fragment (16x32 bf16) straight from repacked weights ----
      FragAB A;
      {
        const unsigned short* wrow =
            wp + (size_t)(co_base + m) * Kt + t * Cin + ci0;
        A.q[0] = *(const uint4*)(wrow + hlf * 16);       // K = half*16 + 0..7
        A.q[1] = *(const uint4*)(wrow + hlf * 16 + 8);   // K = half*16 + 8..15
      }
      // ---- 4 N-subtiles: B frag = contiguous 32B per lane from LDS ----
#pragma unroll
      for (int j = 0; j < 4; ++j) {
        FragAB Bf;
        const unsigned short* bp = &Bs[j * 16 + m + t][hlf * 16];
        Bf.q[0] = *(const uint4*)(bp);
        Bf.q[1] = *(const uint4*)(bp + 8);
        acc[j] = __builtin_amdgcn_wmma_f32_16x16x32_bf16(
            false, A.v, false, Bf.v, (short)0, acc[j], false, false);
      }
    }
  }

  // ---- store raw conv output (bf16), D layout: VGPR r -> M = r + half*8 ----
#pragma unroll
  for (int j = 0; j < 4; ++j) {
    int col = l0 + j * 16 + m;
#pragma unroll
    for (int rr = 0; rr < 8; ++rr) {
      int co = co_base + rr + hlf * 8;
      out[((size_t)co * BB + b) * LL + col] = f2bf(acc[j][rr]);
    }
  }
}

// ---------------- deterministic BN stats: per-block partial sums ----------------
// Async-copies the 8KB chunk into LDS via GLOBAL_LOAD_ASYNC_TO_LDS_B128.
__global__ void stats_part_kernel(const unsigned short* __restrict__ h,
                                  float* __restrict__ part, int chunks) {
  int c = blockIdx.x, chunk = blockIdx.y;
  size_t base = (size_t)c * (BB * LL) + (size_t)chunk * 4096;
  float s = 0.f, s2 = 0.f;
#if HAVE_ASYNC_LDS
  __shared__ __align__(16) unsigned short buf[4096];
#pragma unroll
  for (int i = 0; i < 2; ++i) {
    int e = threadIdx.x * 16 + i * 8;
    async_cp16(h + base + e, &buf[e]);
  }
  async_wait0();
  __syncthreads();
#pragma unroll
  for (int i = 0; i < 16; ++i) {
    float v = bf2f(buf[i * 256 + threadIdx.x]);
    s += v; s2 += v * v;
  }
#else
#pragma unroll
  for (int i = 0; i < 16; ++i) {
    float v = bf2f(h[base + i * 256 + threadIdx.x]);
    s += v; s2 += v * v;
  }
#endif
  for (int off = 16; off > 0; off >>= 1) {
    s  += __shfl_down(s,  off, 32);
    s2 += __shfl_down(s2, off, 32);
  }
  __shared__ float rs[8], rs2[8];
  int w = threadIdx.x >> 5, ln = threadIdx.x & 31;
  if (ln == 0) { rs[w] = s; rs2[w] = s2; }
  __syncthreads();
  if (threadIdx.x == 0) {
    float ts = 0.f, ts2 = 0.f;
    for (int i = 0; i < 8; ++i) { ts += rs[i]; ts2 += rs2[i]; }
    part[((size_t)c * chunks + chunk) * 2 + 0] = ts;
    part[((size_t)c * chunks + chunk) * 2 + 1] = ts2;
  }
}

__global__ void stats_fin_kernel(const float* __restrict__ part,
                                 const float* __restrict__ g, const float* __restrict__ bt,
                                 float* __restrict__ sc, float* __restrict__ sh,
                                 int Cn, int chunks, float invN) {
  int c = blockIdx.x * blockDim.x + threadIdx.x;
  if (c >= Cn) return;
  float s = 0.f, s2 = 0.f;
  for (int i = 0; i < chunks; ++i) {            // fixed order -> deterministic
    s  += part[((size_t)c * chunks + i) * 2 + 0];
    s2 += part[((size_t)c * chunks + i) * 2 + 1];
  }
  float mean = s * invN;
  float var  = s2 * invN - mean * mean;
  float k = g[c] * rsqrtf(var + EPSV);
  sc[c] = k;
  sh[c] = bt[c] - mean * k;
}

// ------------- normalize+relu+mean over L (fused for layer 3) -------------
__global__ void pool_kernel(const unsigned short* __restrict__ h3,
                            const float* __restrict__ sc, const float* __restrict__ sh,
                            float* __restrict__ pooled) {
  int c = blockIdx.x, b = blockIdx.y;
  size_t base = ((size_t)c * BB + b) * LL;
  float scv = sc[c], shv = sh[c], s = 0.f;
#if HAVE_ASYNC_LDS
  __shared__ __align__(16) unsigned short buf[4096];
#pragma unroll
  for (int i = 0; i < 2; ++i) {
    int e = threadIdx.x * 16 + i * 8;
    async_cp16(h3 + base + e, &buf[e]);
  }
  async_wait0();
  __syncthreads();
#pragma unroll
  for (int i = 0; i < 16; ++i)
    s += fmaxf(bf2f(buf[i * 256 + threadIdx.x]) * scv + shv, 0.0f);
#else
  for (int i = threadIdx.x; i < LL; i += blockDim.x)
    s += fmaxf(bf2f(h3[base + i]) * scv + shv, 0.0f);
#endif
  for (int off = 16; off > 0; off >>= 1) s += __shfl_down(s, off, 32);
  __shared__ float rs[8];
  int w = threadIdx.x >> 5, ln = threadIdx.x & 31;
  if (ln == 0) rs[w] = s;
  __syncthreads();
  if (threadIdx.x == 0) {
    float ts = 0.f;
    for (int i = 0; i < 8; ++i) ts += rs[i];
    pooled[b * (4 * CC) + c] = ts * (1.0f / LL);
  }
}

// ---------------- fc + relu ----------------
__global__ void fc_kernel(const float* __restrict__ pooled,
                          const float* __restrict__ fw, const float* __restrict__ fb,
                          float* __restrict__ outp) {
  int idx = blockIdx.x * blockDim.x + threadIdx.x;
  if (idx >= BB * DD) return;
  int b = idx / DD, d = idx % DD;
  const float* pr = pooled + b * (4 * CC);
  const float* wr = fw + d * (4 * CC);
  float s = fb[d];
  for (int i = 0; i < 4 * CC; ++i) s += pr[i] * wr[i];
  outp[idx] = fmaxf(s, 0.0f);
}

// ---------------- gate: softmax + top2 ----------------
__global__ void gate_kernel(const float* __restrict__ fco,
                            const float* __restrict__ gw, const float* __restrict__ gb,
                            float* __restrict__ topv, int* __restrict__ topi) {
  int b = threadIdx.x;
  if (b >= BB) return;
  const float* o = fco + b * DD;
  float lg[NE];
  float mx = -1e30f;
  for (int e = 0; e < NE; ++e) {
    float s = gb[e];
    const float* wr = gw + e * DD;
    for (int d = 0; d < DD; ++d) s += o[d] * wr[d];
    lg[e] = s; mx = fmaxf(mx, s);
  }
  float den = 0.f;
  for (int e = 0; e < NE; ++e) { lg[e] = __expf(lg[e] - mx); den += lg[e]; }
  float inv = 1.0f / den;
  int i0 = 0; float v0 = -1.f;
  for (int e = 0; e < NE; ++e) { float p = lg[e] * inv; if (p > v0) { v0 = p; i0 = e; } }
  int i1 = 0; float v1 = -2.f;
  for (int e = 0; e < NE; ++e) { if (e == i0) continue; float p = lg[e] * inv; if (p > v1) { v1 = p; i1 = e; } }
  topv[b * 2] = v0; topv[b * 2 + 1] = v1;
  topi[b * 2] = i0; topi[b * 2 + 1] = i1;
}

// ---------------- only top-2 experts evaluated ----------------
__global__ void expert_kernel(const float* __restrict__ fco,
                              const float* __restrict__ ew, const float* __restrict__ eb,
                              const float* __restrict__ topv, const int* __restrict__ topi,
                              float* __restrict__ outp) {
  int idx = blockIdx.x * blockDim.x + threadIdx.x;
  if (idx >= BB * NCLS) return;
  int b = idx / NCLS, n = idx % NCLS;
  const float* o = fco + b * DD;
  float acc = 0.f;
#pragma unroll
  for (int k = 0; k < 2; ++k) {
    int e = topi[b * 2 + k];
    const float* wr = ew + ((size_t)e * NCLS + n) * DD;
    float s = eb[e * NCLS + n];
    for (int d = 0; d < DD; ++d) s += o[d] * wr[d];
    acc += topv[b * 2 + k] * s;
  }
  outp[idx] = acc;
}

extern "C" void kernel_launch(void* const* d_in, const int* in_sizes, int n_in,
                              void* d_out, int out_size, void* d_ws, size_t ws_size,
                              hipStream_t stream) {
  const float* x      = (const float*)d_in[0];
  const float* w0     = (const float*)d_in[1];
  const float* g0     = (const float*)d_in[2];
  const float* b0     = (const float*)d_in[3];
  const float* w1     = (const float*)d_in[4];
  const float* g1     = (const float*)d_in[5];
  const float* b1     = (const float*)d_in[6];
  const float* w2     = (const float*)d_in[7];
  const float* g2     = (const float*)d_in[8];
  const float* b2     = (const float*)d_in[9];
  const float* fc_w   = (const float*)d_in[10];
  const float* fc_b   = (const float*)d_in[11];
  const float* gate_w = (const float*)d_in[12];
  const float* gate_b = (const float*)d_in[13];
  const float* exp_w  = (const float*)d_in[14];
  const float* exp_b  = (const float*)d_in[15];
  float* outp = (float*)d_out;

  const int BL = BB * LL;            // 262144
  const int chunks = BL / 4096;      // 64

  size_t off = 0;
  char* base = (char*)d_ws;
  auto alloc = [&](size_t bytes) -> char* {
    char* p = base + off;
    off = (off + bytes + 255) & ~(size_t)255;
    return p;
  };
  unsigned short* xb  = (unsigned short*)alloc((size_t)INC * BB * LL * 2);
  unsigned short* w0p = (unsigned short*)alloc((size_t)CC * INC * 5 * 2);
  unsigned short* w1p = (unsigned short*)alloc((size_t)(2 * CC) * CC * 3 * 2);
  unsigned short* w2p = (unsigned short*)alloc((size_t)(4 * CC) * (2 * CC) * 3 * 2);
  unsigned short* h1  = (unsigned short*)alloc((size_t)CC * BL * 2);
  unsigned short* h2  = (unsigned short*)alloc((size_t)(2 * CC) * BL * 2);
  unsigned short* h3  = (unsigned short*)alloc((size_t)(4 * CC) * BL * 2);
  float* part   = (float*)alloc((size_t)(4 * CC) * chunks * 2 * 4);  // reused per layer
  float* sc1    = (float*)alloc(CC * 4);
  float* sh1    = (float*)alloc(CC * 4);
  float* sc2    = (float*)alloc(2 * CC * 4);
  float* sh2    = (float*)alloc(2 * CC * 4);
  float* sc3    = (float*)alloc(4 * CC * 4);
  float* sh3    = (float*)alloc(4 * CC * 4);
  float* pooled = (float*)alloc((size_t)BB * 4 * CC * 4);
  float* fcout  = (float*)alloc((size_t)BB * DD * 4);
  float* topv   = (float*)alloc(BB * 2 * 4);
  int*   topi   = (int*)alloc(BB * 2 * 4);
  (void)ws_size; (void)n_in; (void)in_sizes; (void)out_size;

  // convert / repack
  convert_x_kernel<<<BB * INC, 256, 0, stream>>>(x, xb);
  repack_w_kernel<<<(CC * INC * 5 + 255) / 256, 256, 0, stream>>>(w0, w0p, CC, INC, 5);
  repack_w_kernel<<<(2 * CC * CC * 3 + 255) / 256, 256, 0, stream>>>(w1, w1p, 2 * CC, CC, 3);
  repack_w_kernel<<<(4 * CC * 2 * CC * 3 + 255) / 256, 256, 0, stream>>>(w2, w2p, 4 * CC, 2 * CC, 3);

  const int ntile = BB * (LL / 64);  // 4096

  // layer 1: conv -> stats -> scale/shift
  conv_wmma_kernel<<<dim3(ntile, CC / 64), 128, 0, stream>>>(
      xb, w0p, h1, nullptr, nullptr, 0, INC, CC, 5, 2);
  stats_part_kernel<<<dim3(CC, chunks), 256, 0, stream>>>(h1, part, chunks);
  stats_fin_kernel<<<(CC + 255) / 256, 256, 0, stream>>>(part, g0, b0, sc1, sh1, CC, chunks, 1.0f / BL);

  // layer 2 (BN+ReLU of h1 fused into staging)
  conv_wmma_kernel<<<dim3(ntile, (2 * CC) / 64), 128, 0, stream>>>(
      h1, w1p, h2, sc1, sh1, 1, CC, 2 * CC, 3, 1);
  stats_part_kernel<<<dim3(2 * CC, chunks), 256, 0, stream>>>(h2, part, chunks);
  stats_fin_kernel<<<(2 * CC + 255) / 256, 256, 0, stream>>>(part, g1, b1, sc2, sh2, 2 * CC, chunks, 1.0f / BL);

  // layer 3
  conv_wmma_kernel<<<dim3(ntile, (4 * CC) / 64), 128, 0, stream>>>(
      h2, w2p, h3, sc2, sh2, 1, 2 * CC, 4 * CC, 3, 1);
  stats_part_kernel<<<dim3(4 * CC, chunks), 256, 0, stream>>>(h3, part, chunks);
  stats_fin_kernel<<<(4 * CC + 255) / 256, 256, 0, stream>>>(part, g2, b2, sc3, sh3, 4 * CC, chunks, 1.0f / BL);

  // fused BN+ReLU+mean pool, then head
  pool_kernel<<<dim3(4 * CC, BB), 256, 0, stream>>>(h3, sc3, sh3, pooled);
  fc_kernel<<<(BB * DD + 255) / 256, 256, 0, stream>>>(pooled, fc_w, fc_b, fcout);
  gate_kernel<<<1, 64, 0, stream>>>(fcout, gate_w, gate_b, topv, topi);
  expert_kernel<<<(BB * NCLS + 255) / 256, 256, 0, stream>>>(fcout, exp_w, exp_b, topv, topi, outp);
}
